// Attention_sep_73744588472674
// MI455X (gfx1250) — matrix operand order repaired and checked by reading, not compiled, not run on hardware
//
#include <hip/hip_runtime.h>
#include <hip/hip_bf16.h>

typedef unsigned short u16;
typedef __attribute__((ext_vector_type(16))) __bf16 v16bf;
typedef __attribute__((ext_vector_type(8)))  float  v8f;

union BF16x16 { v16bf v; u16 u[16]; };

#define NTOK   4301
#define DMODEL 768
#define NHEAD  12
#define HDIM   64
#define NDET_  100
#define NPROP  4201
#define TILES_P 263   // ceil(4201/16)
#define TILES_D 7     // ceil(100/16)
#define ROWT   (TILES_P + TILES_D)   // 270
#define QTILES 269    // ceil(4301/16)
#define SCALE_F 0.125f // 64^-0.5

__device__ __forceinline__ u16 f2bf(float f) {
  unsigned int u = __float_as_uint(f);
  unsigned int r = u + 0x7FFFu + ((u >> 16) & 1u);
  return (u16)(r >> 16);
}

// ---- CDNA5 async memory->LDS staging (ASYNCcnt path, inline asm) ---------
// VGLOBAL encoding (ISA 15.18.3): VDST = per-lane LDS byte address,
// VADDR = 64-bit global address. Tracked by ASYNCcnt.

__device__ __forceinline__ void async_ld_b128(void* lds, const void* g) {
  unsigned int loff = (unsigned int)(unsigned long long)lds;  // LDS byte offset
  unsigned long long ga = (unsigned long long)g;
  asm volatile("global_load_async_to_lds_b128 %0, %1, off"
               :
               : "v"(loff), "v"(ga)
               : "memory");
}

__device__ __forceinline__ void async_wait() {
  asm volatile("s_wait_asynccnt 0x0" ::: "memory");
}

// ---- WMMA helpers (CDNA5 gfx1250, wave32) --------------------------------

__device__ __forceinline__ v8f wmma_bf16(v16bf a, v16bf b, v8f c) {
  return __builtin_amdgcn_wmma_f32_16x16x32_bf16(
      /*neg_a=*/false, a, /*neg_b=*/false, b,
      /*c_mod=*/(short)0, c, /*reuse_a=*/false, /*reuse_b=*/false);
}

// A fragment: 16x32 bf16, LDS holds [m][k] row-major (stride in u16).
__device__ __forceinline__ v16bf load_A(const u16* lds, int stride, int koff) {
  int lane = threadIdx.x & 31;
  int m = lane & 15;
  int kb = (lane >> 4) << 4;
  BF16x16 f;
#pragma unroll
  for (int i = 0; i < 16; ++i) {
    int k = kb + i;
    f.u[i] = lds[m * stride + koff + k];
  }
  return f.v;
}

// B fragment: 32x16 bf16, B[k][n]; LDS holds the TRANSPOSE [n][k] row-major.
__device__ __forceinline__ v16bf load_B_trans(const u16* lds, int stride,
                                              int noff, int koff) {
  int lane = threadIdx.x & 31;
  int n = lane & 15;
  int kb = (lane >> 4) << 4;
  BF16x16 f;
#pragma unroll
  for (int i = 0; i < 16; ++i)
    f.u[i] = lds[(noff + n) * stride + koff + kb + i];
  return f.v;
}

// B fragment where LDS holds B directly as [k][n] row-major.
__device__ __forceinline__ v16bf load_B_direct(const u16* lds, int stride,
                                               int noff) {
  int lane = threadIdx.x & 31;
  int n = lane & 15;
  int kb = (lane >> 4) << 4;
  BF16x16 f;
#pragma unroll
  for (int i = 0; i < 16; ++i)
    f.u[i] = lds[(kb + i) * stride + noff + n];
  return f.v;
}

// ---- f32 -> bf16 conversion ----------------------------------------------

__global__ void cvt_bf16_kernel(const float* __restrict__ in,
                                u16* __restrict__ out, int n) {
  int i = blockIdx.x * blockDim.x + threadIdx.x;
  int st = gridDim.x * blockDim.x;
  for (; i < n; i += st) out[i] = f2bf(in[i]);
}

// ---- QKV projection: one wave computes a 16-row x 64-col (one head) tile --

__global__ __launch_bounds__(32)
void qkv_kernel(const u16* __restrict__ xb,
                const u16* __restrict__ wqp, const u16* __restrict__ wkp,
                const u16* __restrict__ wvp, const u16* __restrict__ wqd,
                const u16* __restrict__ wkd, const u16* __restrict__ wvd,
                const float* __restrict__ bqp, const float* __restrict__ bvp,
                const float* __restrict__ bqd, const float* __restrict__ bvd,
                u16* __restrict__ qb, u16* __restrict__ kb,
                u16* __restrict__ vb) {
  __shared__ __align__(16) u16 a_lds[16 * 32];
  __shared__ __align__(16) u16 w_lds[3][64 * 32];

  const int lane = threadIdx.x;
  const int t = blockIdx.x;
  const int colb = blockIdx.y * HDIM;

  int gstart, row0, nrows;
  const u16 *wq, *wk, *wv;
  const float *bq, *bv;
  if (t < TILES_P) {
    gstart = 0; row0 = t * 16; nrows = NPROP;
    wq = wqp; wk = wkp; wv = wvp; bq = bqp; bv = bvp;
  } else {
    gstart = NPROP; row0 = (t - TILES_P) * 16; nrows = NDET_;
    wq = wqd; wk = wkd; wv = wvd; bq = bqd; bv = bvd;
  }

  v8f accq[4], acck[4], accv[4];
#pragma unroll
  for (int i = 0; i < 4; ++i) {
    accq[i] = (v8f)(0.0f); acck[i] = (v8f)(0.0f); accv[i] = (v8f)(0.0f);
  }

  const u16* wm[3] = {wq, wk, wv};

  for (int k0 = 0; k0 < DMODEL; k0 += 32) {
    // stage X tile [16 rows][32 k] (masked rows -> 0)
    {
      int m = lane >> 1, half = lane & 1;
      int lr = row0 + m;
      if (lr < nrows) {
        const u16* src = xb + (size_t)(gstart + lr) * DMODEL + k0 + half * 16;
        async_ld_b128(&a_lds[m * 32 + half * 16], src);
        async_ld_b128(&a_lds[m * 32 + half * 16 + 8], src + 8);
      } else {
        const uint4 z = make_uint4(0, 0, 0, 0);
        *(uint4*)&a_lds[m * 32 + half * 16] = z;
        *(uint4*)&a_lds[m * 32 + half * 16 + 8] = z;
      }
    }
    // stage W tiles [64 dout][32 k] for q,k,v
#pragma unroll
    for (int w = 0; w < 3; ++w) {
      int j = lane * 2;
      const u16* src = wm[w] + (size_t)(colb + j) * DMODEL + k0;
#pragma unroll
      for (int rr = 0; rr < 2; ++rr)
#pragma unroll
        for (int c = 0; c < 4; ++c)
          async_ld_b128(&w_lds[w][(j + rr) * 32 + c * 8],
                        src + rr * DMODEL + c * 8);
      // hint next k-step tile toward the caches
      if (k0 + 32 < DMODEL) __builtin_prefetch(src + 32, 0, 3);
    }
    async_wait();
    __syncthreads();

    v16bf a = load_A(a_lds, 32, 0);
#pragma unroll
    for (int nt = 0; nt < 4; ++nt) {
      v16bf bfq = load_B_trans(w_lds[0], 32, nt * 16, 0);
      accq[nt] = wmma_bf16(a, bfq, accq[nt]);
      v16bf bfk = load_B_trans(w_lds[1], 32, nt * 16, 0);
      acck[nt] = wmma_bf16(a, bfk, acck[nt]);
      v16bf bfv = load_B_trans(w_lds[2], 32, nt * 16, 0);
      accv[nt] = wmma_bf16(a, bfv, accv[nt]);
    }
    __syncthreads();
  }

  // epilogue: bias, scale, convert, store
  const int hi = (lane >> 4) << 3;
  const int n = lane & 15;
#pragma unroll
  for (int nt = 0; nt < 4; ++nt) {
    int col = colb + nt * 16 + n;
    float bqv = bq[col], bvv = bv[col];
#pragma unroll
    for (int r = 0; r < 8; ++r) {
      int m = r + hi;
      int lr = row0 + m;
      if (lr < nrows) {
        size_t off = (size_t)(gstart + lr) * DMODEL + col;
        qb[off] = f2bf((accq[nt][r] + bqv) * SCALE_F);
        kb[off] = f2bf(acck[nt][r]);
        vb[off] = f2bf(accv[nt][r] + bvv);
      }
    }
  }
}

// ---- Flash attention: one wave per (16 q-rows, head) ---------------------

__global__ __launch_bounds__(32)
void attn_kernel(const u16* __restrict__ qb, const u16* __restrict__ kb,
                 const u16* __restrict__ vb, float* __restrict__ ob) {
  __shared__ __align__(16) u16 q_lds[16 * 64];
  __shared__ __align__(16) u16 k_lds[32 * 64];
  __shared__ __align__(16) u16 v_lds[32 * 64];
  __shared__ __align__(16) float s_lds[16 * 32];
  __shared__ __align__(16) u16 p_lds[16 * 32];
  __shared__ float scale_lds[16];
  __shared__ float inv_lds[16];

  const int lane = threadIdx.x;
  const int h = blockIdx.y;
  const int row0 = blockIdx.x * 16;

  // stage Q tile [16][64]
  {
    int m = lane >> 1, half = lane & 1;
    int g = row0 + m;
    if (g < NTOK) {
      const u16* src = qb + (size_t)g * DMODEL + h * HDIM + half * 32;
#pragma unroll
      for (int c = 0; c < 4; ++c)
        async_ld_b128(&q_lds[m * 64 + half * 32 + c * 8], src + c * 8);
    } else {
      const uint4 z = make_uint4(0, 0, 0, 0);
#pragma unroll
      for (int c = 0; c < 4; ++c)
        *(uint4*)&q_lds[m * 64 + half * 32 + c * 8] = z;
    }
  }
  async_wait();
  __syncthreads();
  const v16bf aq0 = load_A(q_lds, 64, 0);
  const v16bf aq1 = load_A(q_lds, 64, 32);

  float run_max = -1e30f, run_sum = 0.0f;
  v8f o[4];
#pragma unroll
  for (int i = 0; i < 4; ++i) o[i] = (v8f)(0.0f);

  const int nkt = (NTOK + 31) / 32;  // 135
  for (int kt = 0; kt < nkt; ++kt) {
    const int j0 = kt * 32;
    // stage K and V tiles [32 keys][64 d]
    {
      int g = j0 + lane;
      if (g < NTOK) {
        const u16* ks = kb + (size_t)g * DMODEL + h * HDIM;
        const u16* vs = vb + (size_t)g * DMODEL + h * HDIM;
#pragma unroll
        for (int c = 0; c < 8; ++c) {
          async_ld_b128(&k_lds[lane * 64 + c * 8], ks + c * 8);
          async_ld_b128(&v_lds[lane * 64 + c * 8], vs + c * 8);
        }
      } else {
        const uint4 z = make_uint4(0, 0, 0, 0);
#pragma unroll
        for (int c = 0; c < 8; ++c) {
          *(uint4*)&k_lds[lane * 64 + c * 8] = z;
          *(uint4*)&v_lds[lane * 64 + c * 8] = z;
        }
      }
    }
    async_wait();
    __syncthreads();

    // S = Q @ K^T  (two 16x16 j-subtiles, HD=64 -> two K-chunks each)
#pragma unroll
    for (int js = 0; js < 2; ++js) {
      v8f s = (v8f)(0.0f);
      v16bf b0 = load_B_trans(k_lds, 64, js * 16, 0);
      s = wmma_bf16(aq0, b0, s);
      v16bf b1 = load_B_trans(k_lds, 64, js * 16, 32);
      s = wmma_bf16(aq1, b1, s);
      int hi = (lane >> 4) << 3, n = lane & 15;
#pragma unroll
      for (int r = 0; r < 8; ++r)
        s_lds[(r + hi) * 32 + js * 16 + n] = s[r];
    }
    __syncthreads();

    // online softmax across all 32 lanes: row = lane&15, col-half = lane>>4
    {
      int m = lane & 15;
      int cbase = (lane >> 4) << 4;  // 0 or 16
      float mx = -1e30f;
#pragma unroll
      for (int j = 0; j < 16; ++j) {
        int gj = j0 + cbase + j;
        float sv = (gj < NTOK) ? s_lds[m * 32 + cbase + j] : -1e30f;
        mx = fmaxf(mx, sv);
      }
      mx = fmaxf(mx, __shfl_xor(mx, 16, 32));     // merge the two halves
      float nm = fmaxf(run_max, mx);
      float sc = __expf(run_max - nm);
      float ps = 0.0f;
#pragma unroll
      for (int j = 0; j < 16; ++j) {
        int gj = j0 + cbase + j;
        float p = (gj < NTOK) ? __expf(s_lds[m * 32 + cbase + j] - nm) : 0.0f;
        ps += p;
        p_lds[m * 32 + cbase + j] = f2bf(p);
      }
      ps += __shfl_xor(ps, 16, 32);
      run_sum = run_sum * sc + ps;
      run_max = nm;
      if (cbase == 0) scale_lds[m] = sc;
    }
    __syncthreads();

    // rescale O accumulators
    {
      int hi = (lane >> 4) << 3;
#pragma unroll
      for (int dc = 0; dc < 4; ++dc)
#pragma unroll
        for (int r = 0; r < 8; ++r) o[dc][r] *= scale_lds[r + hi];
    }

    // O += P @ V
    v16bf ap = load_A(p_lds, 32, 0);
#pragma unroll
    for (int dc = 0; dc < 4; ++dc) {
      v16bf bv16 = load_B_direct(v_lds, 64, dc * 16);
      o[dc] = wmma_bf16(ap, bv16, o[dc]);
    }
    __syncthreads();
  }

  if (lane < 16) inv_lds[lane] = 1.0f / run_sum;
  __syncthreads();

  {
    int hi = (lane >> 4) << 3, n = lane & 15;
#pragma unroll
    for (int dc = 0; dc < 4; ++dc)
#pragma unroll
      for (int r = 0; r < 8; ++r) {
        int m = r + hi;
        int g = row0 + m;
        if (g < NTOK)
          ob[(size_t)g * DMODEL + h * HDIM + dc * 16 + n] = o[dc][r] * inv_lds[m];
      }
  }
}

// ---- LayerNorm -> bf16 ---------------------------------------------------

__global__ __launch_bounds__(256)
void ln_kernel(const float* __restrict__ ob, const float* __restrict__ g,
               const float* __restrict__ b, u16* __restrict__ nb) {
  __shared__ float red[256];
  const int row = blockIdx.x;
  const float* x = ob + (size_t)row * DMODEL;
  const int tid = threadIdx.x;

  float s = 0.0f;
  for (int c = tid; c < DMODEL; c += 256) s += x[c];
  red[tid] = s;
  __syncthreads();
  for (int st = 128; st > 0; st >>= 1) {
    if (tid < st) red[tid] += red[tid + st];
    __syncthreads();
  }
  const float mu = red[0] / (float)DMODEL;
  __syncthreads();

  float v = 0.0f;
  for (int c = tid; c < DMODEL; c += 256) {
    float d = x[c] - mu;
    v += d * d;
  }
  red[tid] = v;
  __syncthreads();
  for (int st = 128; st > 0; st >>= 1) {
    if (tid < st) red[tid] += red[tid + st];
    __syncthreads();
  }
  const float rstd = rsqrtf(red[0] / (float)DMODEL + 1e-5f);

  for (int c = tid; c < DMODEL; c += 256) {
    float y = (x[c] - mu) * rstd * g[c] + b[c];
    nb[(size_t)row * DMODEL + c] = f2bf(y);
  }
}

// ---- Output projection (region-split), f32 output ------------------------

__global__ __launch_bounds__(32)
void oproj_kernel(const u16* __restrict__ nb, const u16* __restrict__ wop,
                  const u16* __restrict__ wod, const float* __restrict__ bop,
                  const float* __restrict__ bod, float* __restrict__ out) {
  __shared__ __align__(16) u16 a_lds[16 * 32];
  __shared__ __align__(16) u16 w_lds[64 * 32];

  const int lane = threadIdx.x;
  const int t = blockIdx.x;
  const int colb = blockIdx.y * 64;

  int gstart, row0, nrows;
  const u16* w;
  const float* bo;
  if (t < TILES_P) {
    gstart = 0; row0 = t * 16; nrows = NPROP; w = wop; bo = bop;
  } else {
    gstart = NPROP; row0 = (t - TILES_P) * 16; nrows = NDET_; w = wod; bo = bod;
  }

  v8f acc[4];
#pragma unroll
  for (int i = 0; i < 4; ++i) acc[i] = (v8f)(0.0f);

  for (int k0 = 0; k0 < DMODEL; k0 += 32) {
    {
      int m = lane >> 1, half = lane & 1;
      int lr = row0 + m;
      if (lr < nrows) {
        const u16* src = nb + (size_t)(gstart + lr) * DMODEL + k0 + half * 16;
        async_ld_b128(&a_lds[m * 32 + half * 16], src);
        async_ld_b128(&a_lds[m * 32 + half * 16 + 8], src + 8);
      } else {
        const uint4 z = make_uint4(0, 0, 0, 0);
        *(uint4*)&a_lds[m * 32 + half * 16] = z;
        *(uint4*)&a_lds[m * 32 + half * 16 + 8] = z;
      }
    }
    {
      int j = lane * 2;
      const u16* src = w + (size_t)(colb + j) * DMODEL + k0;
#pragma unroll
      for (int rr = 0; rr < 2; ++rr)
#pragma unroll
        for (int c = 0; c < 4; ++c)
          async_ld_b128(&w_lds[(j + rr) * 32 + c * 8],
                        src + rr * DMODEL + c * 8);
      if (k0 + 32 < DMODEL) __builtin_prefetch(src + 32, 0, 3);
    }
    async_wait();
    __syncthreads();

    v16bf a = load_A(a_lds, 32, 0);
#pragma unroll
    for (int nt = 0; nt < 4; ++nt) {
      v16bf bfw = load_B_trans(w_lds, 32, nt * 16, 0);
      acc[nt] = wmma_bf16(a, bfw, acc[nt]);
    }
    __syncthreads();
  }

  const int hi = (lane >> 4) << 3;
  const int n = lane & 15;
#pragma unroll
  for (int nt = 0; nt < 4; ++nt) {
    int col = colb + nt * 16 + n;
    float bv = bo[col];
#pragma unroll
    for (int r = 0; r < 8; ++r) {
      int m = r + hi;
      int lr = row0 + m;
      if (lr < nrows)
        out[(size_t)(gstart + lr) * DMODEL + col] = acc[nt][r] + bv;
    }
  }
}

// ---- Host-side launch ----------------------------------------------------

extern "C" void kernel_launch(void* const* d_in, const int* in_sizes, int n_in,
                              void* d_out, int out_size, void* d_ws,
                              size_t ws_size, hipStream_t stream) {
  (void)in_sizes; (void)n_in; (void)out_size; (void)ws_size;

  const float* x    = (const float*)d_in[0];
  const float* wq_p = (const float*)d_in[1];
  const float* wk_p = (const float*)d_in[2];
  const float* wv_p = (const float*)d_in[3];
  const float* wq_d = (const float*)d_in[4];
  const float* wk_d = (const float*)d_in[5];
  const float* wv_d = (const float*)d_in[6];
  const float* bq_p = (const float*)d_in[7];
  const float* bv_p = (const float*)d_in[8];
  const float* bq_d = (const float*)d_in[9];
  const float* bv_d = (const float*)d_in[10];
  const float* ln_g = (const float*)d_in[11];
  const float* ln_b = (const float*)d_in[12];
  const float* wo_p = (const float*)d_in[13];
  const float* bo_p = (const float*)d_in[14];
  const float* wo_d = (const float*)d_in[15];
  const float* bo_d = (const float*)d_in[16];
  float* out = (float*)d_out;

  char* ws = (char*)d_ws;
  auto alloc = [&](size_t bytes) -> char* {
    char* p = ws;
    ws += (bytes + 255) & ~(size_t)255;
    return p;
  };
  const size_t NX = (size_t)NTOK * DMODEL;
  const size_t NW = (size_t)DMODEL * DMODEL;

  u16* xb   = (u16*)alloc(NX * 2);
  u16* wqpb = (u16*)alloc(NW * 2);
  u16* wkpb = (u16*)alloc(NW * 2);
  u16* wvpb = (u16*)alloc(NW * 2);
  u16* wqdb = (u16*)alloc(NW * 2);
  u16* wkdb = (u16*)alloc(NW * 2);
  u16* wvdb = (u16*)alloc(NW * 2);
  u16* wopb = (u16*)alloc(NW * 2);
  u16* wodb = (u16*)alloc(NW * 2);
  u16* qb   = (u16*)alloc(NX * 2);
  u16* kb   = (u16*)alloc(NX * 2);
  u16* vb   = (u16*)alloc(NX * 2);
  float* ob = (float*)alloc(NX * 4);
  u16* nbuf = (u16*)alloc(NX * 2);

  // f32 -> bf16 conversions
  {
    int gx = (int)((NX + 255) / 256);
    int gw = (int)((NW + 255) / 256);
    cvt_bf16_kernel<<<gx, 256, 0, stream>>>(x, xb, (int)NX);
    cvt_bf16_kernel<<<gw, 256, 0, stream>>>(wq_p, wqpb, (int)NW);
    cvt_bf16_kernel<<<gw, 256, 0, stream>>>(wk_p, wkpb, (int)NW);
    cvt_bf16_kernel<<<gw, 256, 0, stream>>>(wv_p, wvpb, (int)NW);
    cvt_bf16_kernel<<<gw, 256, 0, stream>>>(wq_d, wqdb, (int)NW);
    cvt_bf16_kernel<<<gw, 256, 0, stream>>>(wk_d, wkdb, (int)NW);
    cvt_bf16_kernel<<<gw, 256, 0, stream>>>(wv_d, wvdb, (int)NW);
    cvt_bf16_kernel<<<gw, 256, 0, stream>>>(wo_p, wopb, (int)NW);
    cvt_bf16_kernel<<<gw, 256, 0, stream>>>(wo_d, wodb, (int)NW);
  }

  qkv_kernel<<<dim3(ROWT, NHEAD), 32, 0, stream>>>(
      xb, wqpb, wkpb, wvpb, wqdb, wkdb, wvdb,
      bq_p, bv_p, bq_d, bv_d, qb, kb, vb);

  attn_kernel<<<dim3(QTILES, NHEAD), 32, 0, stream>>>(qb, kb, vb, ob);

  ln_kernel<<<NTOK, 256, 0, stream>>>(ob, ln_g, ln_b, nbuf);

  oproj_kernel<<<dim3(ROWT, NHEAD), 32, 0, stream>>>(
      nbuf, wopb, wodb, bo_p, bo_d, out);
}
